// ICLAttention_77103252898221
// MI455X (gfx1250) — matrix operand order repaired and checked by reading, not compiled, not run on hardware
//
#include <hip/hip_runtime.h>
#include <hip/hip_bf16.h>
#include <cstddef>
#include <cstdint>

typedef __attribute__((ext_vector_type(16))) __bf16 v16bf;
typedef __attribute__((ext_vector_type(8)))  float  v8f;

#define DEV __device__ __forceinline__

DEV unsigned short f32_to_bf16(float f) {
  unsigned int u = __float_as_uint(f);
  u += 0x7FFFu + ((u >> 16) & 1u);          // round-to-nearest-even
  return (unsigned short)(u >> 16);
}
DEV unsigned int pack_bf16x2(float lo, float hi) {
  return (unsigned int)f32_to_bf16(lo) | ((unsigned int)f32_to_bf16(hi) << 16);
}

union Frag16 {
  v16bf v;
  unsigned int w[8];
  unsigned short h[16];
};

DEV v8f vzero() { v8f z = {}; return z; }

// 16-byte load (global or LDS) into 4 dwords of a fragment
DEV void ld16(unsigned int* d, const unsigned short* p) {
  const uint4 t = *reinterpret_cast<const uint4*>(p);
  d[0] = t.x; d[1] = t.y; d[2] = t.z; d[3] = t.w;
}

// wave-private LDS producer->consumer ordering (plus compiler fence)
DEV void lds_fence() { asm volatile("s_wait_dscnt 0" ::: "memory"); }

// CDNA5 async memory->LDS copy, 16B per lane, tracked by ASYNCcnt.
// lds_off = wave-relative LDS byte offset (low 32 bits of a generic
// pointer into __shared__), gaddr = per-lane 64-bit global address.
DEV void async_copy16(unsigned lds_off, const void* gaddr) {
  asm volatile("global_load_async_to_lds_b128 %0, %1, off"
               :: "v"(lds_off), "v"(gaddr) : "memory");
}
DEV void async_fence() { asm volatile("s_wait_asynccnt 0x0" ::: "memory"); }

// ---------------------------------------------------------------------------
// GEMM: C[M,N] = A[M,K] @ B[K,N], bf16 WMMA math, f32 accumulate.
// Block = 128 threads (4 waves), 64x64 tile, BK=32.
// Wave w computes rows [16w,16w+16) x all 64 cols (4 C fragments).
// A_BF16 path stages the A tile with async LDS copies (pure bf16 copy).
// ---------------------------------------------------------------------------
template<bool A_BF16, bool OUT_BF16>
__global__ __launch_bounds__(128)
void gemm_wmma_64x64(const void* __restrict__ Ap, const float* __restrict__ Bp,
                     void* __restrict__ Cp, int K, int lda, int ldb, int ldc)
{
  __shared__ unsigned short As[64 * 40];   // [m][k], stride 40 (80B, 16B-aligned)
  __shared__ unsigned short BsT[64 * 40];  // [n][k] transposed, stride 40

  const int tid  = threadIdx.x;
  const int lane = tid & 31;
  const int wv   = tid >> 5;
  const int row0 = blockIdx.y * 64;
  const int col0 = blockIdx.x * 64;
  const int m0   = wv * 16;

  const int nloc = lane & 15;
  const int kbA  = (lane >> 4) << 3;   // A frag K sub-base: 0 / 8
  const int kbB  = (lane >> 4) << 4;   // unused (B frag uses kbA: 0 / 8)

  v8f acc[4] = {};

  for (int k0 = 0; k0 < K; k0 += 32) {
    // ---- stage A tile 64x32 -> bf16 LDS ----
    if constexpr (A_BF16) {
      const unsigned short* Ag = (const unsigned short*)Ap;
#pragma unroll
      for (int i = 0; i < 2; ++i) {            // 2 x 16B per thread, async
        int e = (i * 128 + tid) * 8;           // element index
        int r = e >> 5, c = e & 31;            // c in {0,8,16,24}
        async_copy16((unsigned)(uintptr_t)&As[r * 40 + c],
                     Ag + (size_t)(row0 + r) * lda + k0 + c);
      }
    } else {
      const float* Ag = (const float*)Ap;
#pragma unroll
      for (int i = 0; i < 4; ++i) {
        int e = i * 512 + tid * 4;
        int r = e >> 5, c = e & 31;
        const float4 vv = *reinterpret_cast<const float4*>(
            Ag + (size_t)(row0 + r) * lda + k0 + c);
        unsigned int* q = reinterpret_cast<unsigned int*>(&As[r * 40 + c]);
        q[0] = pack_bf16x2(vv.x, vv.y);
        q[1] = pack_bf16x2(vv.z, vv.w);
      }
    }
    // ---- stage B tile 32x64, transposed into BsT[n][k] ----
#pragma unroll
    for (int i = 0; i < 4; ++i) {
      int e = i * 512 + tid * 4;
      int kk = e >> 6, n = e & 63;
      const float4 vv = *reinterpret_cast<const float4*>(
          Bp + (size_t)(k0 + kk) * ldb + col0 + n);
      BsT[(n + 0) * 40 + kk] = f32_to_bf16(vv.x);
      BsT[(n + 1) * 40 + kk] = f32_to_bf16(vv.y);
      BsT[(n + 2) * 40 + kk] = f32_to_bf16(vv.z);
      BsT[(n + 3) * 40 + kk] = f32_to_bf16(vv.w);
    }
    if constexpr (A_BF16) async_fence();   // async A writes visible before barrier
    __syncthreads();

    // A fragment: lane holds row m0+nloc; K = kbA..kbA+7 and 16+kbA..16+kbA+7
    Frag16 a;
    ld16(a.w,     &As[(m0 + nloc) * 40 + kbA]);
    ld16(a.w + 4, &As[(m0 + nloc) * 40 + 16 + kbA]);

#pragma unroll
    for (int cn = 0; cn < 4; ++cn) {
      // B fragment: lane holds col cn*16+nloc; K = kbA..kbA+7 and 16+kbA..16+kbA+7
      Frag16 bfr;
      ld16(bfr.w,     &BsT[(cn * 16 + nloc) * 40 + kbA]);
      ld16(bfr.w + 4, &BsT[(cn * 16 + nloc) * 40 + 16 + kbA]);
      acc[cn] = __builtin_amdgcn_wmma_f32_16x16x32_bf16(
          false, a.v, false, bfr.v, (short)0, acc[cn], false, false);
    }
    __syncthreads();
  }

  // C layout: lane<16 -> rows 0..7, lane>=16 -> rows 8..15; col = nloc
  const int mbase = row0 + m0 + ((lane >> 4) << 3);
#pragma unroll
  for (int cn = 0; cn < 4; ++cn) {
    int n = col0 + cn * 16 + nloc;
#pragma unroll
    for (int r = 0; r < 8; ++r) {
      float v = acc[cn][r];
      if (OUT_BF16)
        ((unsigned short*)Cp)[(size_t)(mbase + r) * ldc + n] = f32_to_bf16(v);
      else
        ((float*)Cp)[(size_t)(mbase + r) * ldc + n] = v;
    }
  }
}

// ---------------------------------------------------------------------------
// Flash attention, one wave per (b, h, 16-query-row block).
// Near-causal mask: allowed iff (j < i) || (i==0 && j==0). scale = 1/32.
// Q/K/V/O in bf16 [B*S, 1024], head h uses column slice [64h, 64h+64).
// V tile is staged with async LDS copies; the ASYNCcnt wait sits after the
// score WMMAs + softmax, so the copy overlaps with compute.
// ---------------------------------------------------------------------------
__global__ __launch_bounds__(128)
void attn_fmha(const unsigned short* __restrict__ Qb,
               const unsigned short* __restrict__ Kb,
               const unsigned short* __restrict__ Vb,
               unsigned short* __restrict__ Ob)
{
  constexpr int S = 2048, HF = 1024, DH = 64;
  constexpr float SCALE = 0.03125f;  // 1/sqrt(1024)
  constexpr float NEG   = -1e30f;

  __shared__ float          sS[4][16][33];   // per-wave score tile (f32)
  __shared__ unsigned short sP[4][16][40];   // per-wave probs (bf16), 16B rows
  __shared__ unsigned short sV[4][32][64];   // per-wave V tile (bf16)
  __shared__ float          sAlpha[4][16];
  __shared__ float          sLinv[4][16];

  const int lane = threadIdx.x & 31;
  const int wv   = threadIdx.x >> 5;
  const int g    = blockIdx.x * 4 + wv;      // 4096 waves total
  const int q0   = (g & 127) * 16;           // S/16 = 128 blocks
  const int h    = (g >> 7) & 15;
  const int b    = g >> 11;

  const int nloc = lane & 15;
  const int kbA  = (lane >> 4) << 3;
  const int kbB  = (lane >> 4) << 4;
  const int mb   = (lane >> 4) << 3;         // C-frag row base (0 or 8)

  // Q fragments, kept in registers for the whole row-block (16x64 -> 2 A-frags)
  Frag16 aq[2];
  {
    const unsigned short* qrow = Qb + (size_t)(b * S + q0 + nloc) * HF + h * DH;
#pragma unroll
    for (int ck = 0; ck < 2; ++ck) {
      ld16(aq[ck].w,     qrow + ck * 32 + kbA);
      ld16(aq[ck].w + 4, qrow + ck * 32 + 16 + kbA);
    }
  }

  v8f o[4] = {};
  float m_run = NEG, l_run = 0.f;
  const int row = lane >> 1, half = lane & 1;   // softmax: 2 lanes per row

  for (int t0 = 0; t0 < q0 + 16; t0 += 32) {
    // ---- async-stage V tile (32 keys x 64 d): lane copies key row t0+lane ----
    {
      const unsigned short* vrow = Vb + (size_t)(b * S + t0 + lane) * HF + h * DH;
      const unsigned dst = (unsigned)(uintptr_t)&sV[wv][lane][0];
#pragma unroll
      for (int c = 0; c < 8; ++c)
        async_copy16(dst + c * 16, vrow + c * 8);
    }
    // prefetch next K tile (one row per lane) into cache
    if (t0 + 32 < q0 + 16)
      __builtin_prefetch(Kb + (size_t)(b * S + t0 + 32 + lane) * HF + h * DH, 0, 3);

    // ---- scores 16x32 = Q . K^T, straight from global K (B-layout reads) ----
#pragma unroll
    for (int nh = 0; nh < 2; ++nh) {
      v8f s = vzero();
#pragma unroll
      for (int ck = 0; ck < 2; ++ck) {
        Frag16 bk;  // B[k=d][n=key]: lane holds key t0+16nh+nloc, d contiguous
        const unsigned short* krow =
            Kb + (size_t)(b * S + t0 + nh * 16 + nloc) * HF + h * DH + ck * 32 + kbA;
        ld16(bk.w, krow);
        ld16(bk.w + 4, krow + 16);
        s = __builtin_amdgcn_wmma_f32_16x16x32_bf16(
            false, aq[ck].v, false, bk.v, (short)0, s, false, false);
      }
      // mask + scale while spilling to LDS
#pragma unroll
      for (int r = 0; r < 8; ++r) {
        int i  = q0 + mb + r;
        int jl = nh * 16 + nloc;
        int j  = t0 + jl;
        bool ok = (j < i) || ((i == 0) && (j == 0));
        sS[wv][mb + r][jl] = ok ? s[r] * SCALE : NEG;
      }
    }
    lds_fence();

    // ---- online softmax: lane pair (row, half) scans 16 cols each ----
    float tmax = NEG;
    float svv[16];
#pragma unroll
    for (int e = 0; e < 16; ++e) {
      svv[e] = sS[wv][row][half * 16 + e];
      tmax = fmaxf(tmax, svv[e]);
    }
    tmax = fmaxf(tmax, __shfl_xor(tmax, 1));
    float mnew  = fmaxf(m_run, tmax);
    float alpha = __expf(m_run - mnew);
    float lsum  = 0.f;
#pragma unroll
    for (int e = 0; e < 16; ++e) {
      float p = (svv[e] < -9e29f) ? 0.f : __expf(svv[e] - mnew);
      lsum += p;
      sP[wv][row][half * 16 + e] = f32_to_bf16(p);
    }
    lsum += __shfl_xor(lsum, 1);
    l_run = l_run * alpha + lsum;
    m_run = mnew;
    if (half == 0) sAlpha[wv][row] = alpha;
    lds_fence();

    // ---- rescale O, then O += P @ V ----
#pragma unroll
    for (int r = 0; r < 8; ++r) {
      float a = sAlpha[wv][mb + r];
#pragma unroll
      for (int cn = 0; cn < 4; ++cn) o[cn][r] *= a;
    }

    Frag16 ap;  // P as A-fragment (16x32)
    ld16(ap.w,     &sP[wv][nloc][kbA]);
    ld16(ap.w + 4, &sP[wv][nloc][16 + kbA]);

    async_fence();   // V tile async copies must have landed in LDS by now

#pragma unroll
    for (int cn = 0; cn < 4; ++cn) {
      Frag16 bv;  // B[k=key][n=d]: lane holds d = cn*16+nloc, keys kbA..kbA+7, 16+kbA..16+kbA+7
#pragma unroll
      for (int e = 0; e < 16; ++e)
        bv.h[e] = sV[wv][(e < 8 ? kbA + e : 16 + kbA + (e - 8))][cn * 16 + nloc];
      o[cn] = __builtin_amdgcn_wmma_f32_16x16x32_bf16(
          false, ap.v, false, bv.v, (short)0, o[cn], false, false);
    }
    lds_fence();  // gathers done before next tile's async copies overwrite sV
  }

  if (half == 0) sLinv[wv][row] = 1.0f / l_run;
  lds_fence();

#pragma unroll
  for (int cn = 0; cn < 4; ++cn) {
#pragma unroll
    for (int r = 0; r < 8; ++r) {
      float li = sLinv[wv][mb + r];
      Ob[(size_t)(b * S + q0 + mb + r) * HF + h * DH + cn * 16 + nloc] =
          f32_to_bf16(o[cn][r] * li);
    }
  }
}

// ---------------------------------------------------------------------------
extern "C" void kernel_launch(void* const* d_in, const int* in_sizes, int n_in,
                              void* d_out, int out_size, void* d_ws, size_t ws_size,
                              hipStream_t stream)
{
  (void)in_sizes; (void)n_in; (void)out_size; (void)ws_size;

  const float* q  = (const float*)d_in[0];
  const float* k  = (const float*)d_in[1];
  const float* v  = (const float*)d_in[2];
  const float* Wq = (const float*)d_in[3];
  const float* Wk = (const float*)d_in[4];
  const float* Wv = (const float*)d_in[5];
  const float* Wo = (const float*)d_in[6];
  float* out = (float*)d_out;

  const int M = 2 * 2048;   // B*S
  const int K = 1024;
  const int N = 1024;

  unsigned short* Qb = (unsigned short*)d_ws;           // bf16 [M,N]
  unsigned short* Kb = Qb + (size_t)M * N;
  unsigned short* Vb = Kb + (size_t)M * N;
  unsigned short* Ob = Vb + (size_t)M * N;              // 32 MB total

  dim3 gg(N / 64, M / 64);  // (16, 64)
  gemm_wmma_64x64<false, true><<<gg, 128, 0, stream>>>(q, Wq, Qb, K, K, N, N);
  gemm_wmma_64x64<false, true><<<gg, 128, 0, stream>>>(k, Wk, Kb, K, K, N, N);
  gemm_wmma_64x64<false, true><<<gg, 128, 0, stream>>>(v, Wv, Vb, K, K, N, N);

  attn_fmha<<<1024, 128, 0, stream>>>(Qb, Kb, Vb, Ob);

  gemm_wmma_64x64<true, false><<<gg, 128, 0, stream>>>(Ob, Wo, out, K, K, N, N);
}
